// DeltaNet_31877247271528
// MI455X (gfx1250) — matrix operand order repaired and checked by reading, not compiled, not run on hardware
//
#include <hip/hip_runtime.h>
#include <math.h>

// ---------------------------------------------------------------------------
// Types for CDNA5 WMMA (wave32, 16x16x32 bf16 -> f32)
// ---------------------------------------------------------------------------
typedef __attribute__((ext_vector_type(16))) __bf16 bf16x16;
typedef __attribute__((ext_vector_type(8)))  float  f32x8;

__device__ __forceinline__ int lane_id() { return threadIdx.x & 31; }

__device__ __forceinline__ f32x8 zero8() {
    f32x8 z;
#pragma unroll
    for (int e = 0; e < 8; ++e) z[e] = 0.f;
    return z;
}

__device__ __forceinline__ f32x8 wmma_bf16(bf16x16 a, bf16x16 b, f32x8 c) {
    return __builtin_amdgcn_wmma_f32_16x16x32_bf16(false, a, false, b, (short)0, c,
                                                   false, false);
}

// A fragment: matrix stored row-major [M][K], p points at tile row m0, col 0.
// Contiguous k order used here (same as B loaders): lane l -> m = l&15, g = l>>4; elem e: K = 16g + e
__device__ __forceinline__ bf16x16 ld_a(const __bf16* p, int ld, int kb) {
    int l = lane_id(); int m = l & 15; int g = l >> 4;
    const __bf16* r0 = p + (size_t)m * ld + kb + g * 16;
    const __bf16* r1 = r0 + 8;
    bf16x16 v;
#pragma unroll
    for (int e = 0; e < 8; ++e) { v[e] = r0[e]; v[e + 8] = r1[e]; }
    return v;
}

// A fragment of X^T where X is stored row-major [K][M]: element (m,k)=X[k][m]
__device__ __forceinline__ bf16x16 ld_aT(const __bf16* p, int ld, int m0, int kb) {
    int l = lane_id(); int m = m0 + (l & 15); int g = l >> 4;
    bf16x16 v;
#pragma unroll
    for (int e = 0; e < 8; ++e) {
        v[e]     = p[(size_t)(kb + 16 * g + e) * ld + m];
        v[e + 8] = p[(size_t)(kb + 16 * g + 8 + e) * ld + m];
    }
    return v;
}

// B fragment for W stored row-major [N][K] (i.e. computing A @ W^T).
// Contiguous k order used here (same as ld_a): lane l -> n = l&15, g = l>>4; elem e: K = 16g + e
__device__ __forceinline__ bf16x16 ld_b(const __bf16* p, int ld, int kb) {
    int l = lane_id(); int n = l & 15; int g = l >> 4;
    const __bf16* r = p + (size_t)n * ld + kb + g * 16;
    bf16x16 v;
#pragma unroll
    for (int e = 0; e < 16; ++e) v[e] = r[e];
    return v;
}

// B fragment for X stored row-major [K][N]
__device__ __forceinline__ bf16x16 ld_b_kxn(const __bf16* p, int ld, int n0, int kb) {
    int l = lane_id(); int n = n0 + (l & 15); int g = l >> 4;
    const __bf16* base = p + (size_t)(kb + g * 16) * ld + n;
    bf16x16 v;
#pragma unroll
    for (int e = 0; e < 16; ++e) v[e] = base[(size_t)e * ld];
    return v;
}

// Same but source is f32 (convert on the fly) -- used to read S from LDS
__device__ __forceinline__ bf16x16 ld_b_kxn_f32(const float* p, int ld, int n0, int kb) {
    int l = lane_id(); int n = n0 + (l & 15); int g = l >> 4;
    const float* base = p + (size_t)(kb + g * 16) * ld + n;
    bf16x16 v;
#pragma unroll
    for (int e = 0; e < 16; ++e) v[e] = (__bf16)base[(size_t)e * ld];
    return v;
}

// C/D fragment helpers. p points at tile origin. Per ISA: VGPR r holds M=r
// (lanes 0-15) / M=r+8 (lanes 16-31); column = lane&15.
__device__ __forceinline__ f32x8 ld_c_f32(const float* p, int ld) {
    int l = lane_id(); int n = l & 15; int g = l >> 4;
    f32x8 c;
#pragma unroll
    for (int r = 0; r < 8; ++r) c[r] = p[(size_t)(r + 8 * g) * ld + n];
    return c;
}
__device__ __forceinline__ void st_c_f32(float* p, int ld, f32x8 c) {
    int l = lane_id(); int n = l & 15; int g = l >> 4;
#pragma unroll
    for (int r = 0; r < 8; ++r) p[(size_t)(r + 8 * g) * ld + n] = c[r];
}
__device__ __forceinline__ void st_c_bf16(__bf16* p, int ld, f32x8 c) {
    int l = lane_id(); int n = l & 15; int g = l >> 4;
#pragma unroll
    for (int r = 0; r < 8; ++r) p[(size_t)(r + 8 * g) * ld + n] = (__bf16)c[r];
}

// ---------------------------------------------------------------------------
// Elementwise / conversion kernels
// ---------------------------------------------------------------------------
__global__ __launch_bounds__(256) void cvt_bf16_kernel(const float* __restrict__ s,
                                                       __bf16* __restrict__ d, int n) {
    int i = blockIdx.x * 256 + threadIdx.x;
    if (i < n) d[i] = (__bf16)s[i];
}

// fuse_w1 (1024 x 1040) -> bf16 (1024 x 1056), zero padded
__global__ __launch_bounds__(256) void cvt_w1_kernel(const float* __restrict__ s,
                                                     __bf16* __restrict__ d, int rows) {
    int i = blockIdx.x * 256 + threadIdx.x;
    int tot = rows * 1056;
    if (i >= tot) return;
    int r = i / 1056, c = i - r * 1056;
    d[i] = (c < 1040) ? (__bf16)s[(size_t)r * 1040 + c] : (__bf16)0.f;
}

// depthwise causal conv (K=4) + SiLU over (B*L, 1024)
__global__ __launch_bounds__(256) void convsilu_kernel(const float* __restrict__ x,
                                                       const float* __restrict__ w,
                                                       float* __restrict__ y, int n) {
    int i = blockIdx.x * 256 + threadIdx.x;
    if (i >= n) return;
    int c = i & 1023; int bl = i >> 10; int l = bl & 4095;
    float acc = 0.f;
#pragma unroll
    for (int t = 0; t < 4; ++t) {
        int ll = l - 3 + t;
        if (ll >= 0) acc += w[c * 4 + t] * x[(size_t)(bl - 3 + t) * 1024 + c];
    }
    y[i] = acc / (1.f + expf(-acc));
}

// depthwise causal FIR (generic K) over (B*L, 1024); filter flat [1024][K]
__global__ __launch_bounds__(256) void fir_kernel(const float* __restrict__ x,
                                                  const float* __restrict__ f,
                                                  float* __restrict__ y, int K, int n) {
    int i = blockIdx.x * 256 + threadIdx.x;
    if (i >= n) return;
    int c = i & 1023; int bl = i >> 10; int l = bl & 4095;
    float acc = 0.f;
    for (int t = 0; t < K; ++t) {
        int ll = l - (K - 1) + t;
        if (ll >= 0) acc += f[(size_t)c * K + t] * x[(size_t)(bl - (K - 1) + t) * 1024 + c];
    }
    y[i] = acc;
}

// beta[bl*4+h] = sigmoid(hs[bl,:] . b_proj_w[h,:]) ; one wave per job
__global__ __launch_bounds__(256) void beta_kernel(const float* __restrict__ hs,
                                                   const float* __restrict__ bw,
                                                   float* __restrict__ beta, int jobs) {
    int j = (blockIdx.x * 256 + threadIdx.x) >> 5;
    int lane = threadIdx.x & 31;
    if (j >= jobs) return;
    int bl = j >> 2, h = j & 3;
    const float* xr = hs + (size_t)bl * 1024;
    const float* wr = bw + (size_t)h * 1024;
    float s = 0.f;
#pragma unroll
    for (int e = 0; e < 32; ++e) { int c = e * 32 + lane; s += xr[c] * wr[c]; }
#pragma unroll
    for (int m = 16; m >= 1; m >>= 1) s += __shfl_xor(s, m, 32);
    if (lane == 0) beta[j] = 1.f / (1.f + expf(-s));
}

// ---------------------------------------------------------------------------
// Generic bf16 WMMA GEMM: C(MxN) = A(MxK) @ Bw(NxK)^T ; optional GeLU epilogue
// mode 0: write f32 to Cf ; mode 1: write bf16 gelu(acc + bias[n]) to Cb
// Requires M%128==0, N%128==0, K%32==0 (guaranteed by construction).
// ---------------------------------------------------------------------------
__global__ __launch_bounds__(256) void gemm_bf16_kernel(
    const __bf16* __restrict__ A, const __bf16* __restrict__ Bw,
    float* __restrict__ Cf, __bf16* __restrict__ Cb, const float* __restrict__ bias,
    int M, int N, int K, int lda, int ldb, int ldc, int mode) {
    __shared__ __bf16 As[128][40];
    __shared__ __bf16 Bs[128][40];
    int tid = threadIdx.x;
    int m0 = blockIdx.y * 128, n0 = blockIdx.x * 128;
    int wid = tid >> 5;
    int wm = wid & 3, wn = wid >> 2;  // 4x2 wave grid; wave tile 32(M) x 64(N)
    f32x8 acc[2][4];
#pragma unroll
    for (int i = 0; i < 2; ++i)
#pragma unroll
        for (int j = 0; j < 4; ++j) acc[i][j] = zero8();

    int r = tid >> 1, sg = (tid & 1) * 16;
    for (int k0 = 0; k0 < K; k0 += 32) {
        const __bf16* ga = A + (size_t)(m0 + r) * lda + k0 + sg;
        const __bf16* gb = Bw + (size_t)(n0 + r) * ldb + k0 + sg;
        if (k0 + 32 < K) {  // warm L2/L0 for the next K tile (global_prefetch_b8)
            __builtin_prefetch(ga + 32, 0, 1);
            __builtin_prefetch(gb + 32, 0, 1);
        }
#pragma unroll
        for (int e = 0; e < 16; ++e) { As[r][sg + e] = ga[e]; Bs[r][sg + e] = gb[e]; }
        __syncthreads();

        bf16x16 af[2];
#pragma unroll
        for (int i = 0; i < 2; ++i) af[i] = ld_a(&As[wm * 32 + i * 16][0], 40, 0);
#pragma unroll
        for (int j = 0; j < 4; ++j) {
            bf16x16 bfrag = ld_b(&Bs[wn * 64 + j * 16][0], 40, 0);
#pragma unroll
            for (int i = 0; i < 2; ++i) acc[i][j] = wmma_bf16(af[i], bfrag, acc[i][j]);
        }
        __syncthreads();
    }

    int l = tid & 31; int cg = l >> 4; int cn = l & 15;
#pragma unroll
    for (int i = 0; i < 2; ++i) {
#pragma unroll
        for (int j = 0; j < 4; ++j) {
            int row0 = m0 + wm * 32 + i * 16;
            int col = n0 + wn * 64 + j * 16 + cn;
#pragma unroll
            for (int rr = 0; rr < 8; ++rr) {
                int row = row0 + rr + 8 * cg;
                float v = acc[i][j][rr];
                if (mode == 1) {
                    v += bias[col];
                    v = 0.5f * v * (1.f + erff(v * 0.70710678118654752f));
                    Cb[(size_t)row * ldc + col] = (__bf16)v;
                } else {
                    Cf[(size_t)row * ldc + col] = v;
                }
            }
        }
    }
}

// ---------------------------------------------------------------------------
// Delta-rule chunkwise scan. One workgroup per (b, h, dv-half).
// State S (dk=256 x dv=128, f32) lives in LDS across the whole scan.
// All matmuls are v_wmma_f32_16x16x32_bf16 with f32 accumulation.
// Dynamic LDS: 212352 bytes.
// ---------------------------------------------------------------------------
__global__ __launch_bounds__(256) void delta_kernel(
    const float* __restrict__ q, const float* __restrict__ k,
    const float* __restrict__ v, const float* __restrict__ beta,
    float* __restrict__ dout) {
    extern __shared__ char sm[];
    float*  S     = (float*)sm;                       // 256 x 132 f32
    __bf16* qb    = (__bf16*)(sm + 256 * 132 * 4);    // 32 x 264
    __bf16* knb   = qb  + 32 * 264;                   // 32 x 264
    __bf16* wb    = knb + 32 * 264;                   // 32 x 264
    __bf16* vb    = wb  + 32 * 264;                   // 32 x 136
    __bf16* uab   = vb  + 32 * 136;                   // 32 x 136
    float*  Amat  = (float*)(uab + 32 * 136);         // 32 x 34
    __bf16* invb  = (__bf16*)(Amat + 32 * 34);        // 32 x 36
    __bf16* attnb = invb + 32 * 36;                   // 32 x 36
    float*  betas = (float*)(attnb + 32 * 36);        // 32

    const int b    = blockIdx.x >> 3;
    const int h    = (blockIdx.x >> 1) & 3;
    const int half = blockIdx.x & 1;
    const int tid = threadIdx.x, wid = tid >> 5, lane = tid & 31;

    for (int i = tid; i < 256 * 132; i += 256) S[i] = 0.f;
    __syncthreads();

    const int mt  = wid & 1;          // M tile (tokens) for 32xN outputs
    const int ntb = (wid >> 1) * 2;   // pair of N tiles over dv=128

    for (int ci = 0; ci < 128; ++ci) {
        const int l0 = ci * 32;

        // --- load chunk, l2-normalize q/k, stage bf16 operands --------------
#pragma unroll
        for (int i = 0; i < 4; ++i) {
            int tok = wid * 4 + i;
            size_t gl = (size_t)(b * 4096 + l0 + tok);
            const float* qr = q + gl * 1024 + h * 256;
            const float* kr = k + gl * 1024 + h * 256;
            float qv[8], kv[8], sq = 0.f, sk = 0.f;
#pragma unroll
            for (int e = 0; e < 8; ++e) {
                int d = e * 32 + lane;
                qv[e] = qr[d]; kv[e] = kr[d];
                sq += qv[e] * qv[e]; sk += kv[e] * kv[e];
            }
#pragma unroll
            for (int m = 16; m >= 1; m >>= 1) {
                sq += __shfl_xor(sq, m, 32);
                sk += __shfl_xor(sk, m, 32);
            }
            float rq = rsqrtf(sq), rk = rsqrtf(sk);
#pragma unroll
            for (int e = 0; e < 8; ++e) {
                int d = e * 32 + lane;
                qb[tok * 264 + d]  = (__bf16)(qv[e] * rq);
                knb[tok * 264 + d] = (__bf16)(kv[e] * rk);
            }
            const float* vr = v + gl * 1024 + h * 256 + half * 128;
#pragma unroll
            for (int e = 0; e < 4; ++e) {
                int d = e * 32 + lane;
                vb[tok * 136 + d] = (__bf16)vr[d];
            }
            if (lane == 0) betas[tok] = beta[gl * 4 + h];
        }
        __syncthreads();

        // --- A = where(j<i, -beta_i * (kn_i . kn_j), 0) ----------------------
        if (wid < 4) {
            int mta = wid >> 1, nta = wid & 1;
            f32x8 acc = zero8();
            for (int k0 = 0; k0 < 256; k0 += 32)
                acc = wmma_bf16(ld_a(knb + mta * 16 * 264, 264, k0),
                                ld_b(knb + nta * 16 * 264, 264, k0), acc);
            int g = lane >> 4, n = lane & 15;
#pragma unroll
            for (int r = 0; r < 8; ++r) {
                int ii = mta * 16 + r + 8 * g, jj = nta * 16 + n;
                Amat[ii * 34 + jj] = (jj < ii) ? (-betas[ii] * acc[r]) : 0.f;
            }
        }
        __syncthreads();

        // --- forward-substitution (single wave, lockstep) --------------------
        if (wid == 0) {
            for (int i = 1; i < 32; ++i) {
                float upd = 0.f;
                for (int t = 0; t < 32; ++t) upd += Amat[i * 34 + t] * Amat[t * 34 + lane];
                if (lane < i) Amat[i * 34 + lane] += upd;
            }
        }
        __syncthreads();
        // invb[i][j] = (inv = A + I)[i][j] * beta_j   (folds beta of kb/v)
        for (int idx = tid; idx < 1024; idx += 256) {
            int ii = idx >> 5, jj = idx & 31;
            float val = (ii == jj) ? 1.f : ((jj < ii) ? Amat[ii * 34 + jj] : 0.f);
            invb[ii * 36 + jj] = (__bf16)(val * betas[jj]);
        }
        __syncthreads();

        // --- u = invb @ v (regs) ; w = invb @ kn (-> wb) ---------------------
        bf16x16 ai = ld_a(invb + mt * 16 * 36, 36, 0);
        f32x8 uacc[2];
#pragma unroll
        for (int j = 0; j < 2; ++j)
            uacc[j] = wmma_bf16(ai, ld_b_kxn(vb, 136, (ntb + j) * 16, 0), zero8());
        {
            int ntb4 = (wid >> 1) * 4;
#pragma unroll
            for (int j = 0; j < 4; ++j) {
                f32x8 wacc = wmma_bf16(ai, ld_b_kxn(knb, 264, (ntb4 + j) * 16, 0), zero8());
                st_c_bf16(wb + mt * 16 * 264 + (ntb4 + j) * 16, 264, wacc);
            }
        }
        __syncthreads();

        // --- u_adj = u - w @ S  (-> uab bf16) --------------------------------
        f32x8 ws[2]; ws[0] = zero8(); ws[1] = zero8();
        for (int k0 = 0; k0 < 256; k0 += 32) {
            bf16x16 aw = ld_a(wb + mt * 16 * 264, 264, k0);
#pragma unroll
            for (int j = 0; j < 2; ++j)
                ws[j] = wmma_bf16(aw, ld_b_kxn_f32(S, 132, (ntb + j) * 16, k0), ws[j]);
        }
#pragma unroll
        for (int j = 0; j < 2; ++j) {
            f32x8 ua;
#pragma unroll
            for (int e = 0; e < 8; ++e) ua[e] = uacc[j][e] - ws[j][e];
            st_c_bf16(uab + mt * 16 * 136 + (ntb + j) * 16, 136, ua);
        }
        __syncthreads();

        // --- attn = where(j<=i, q @ kn^T, 0) ---------------------------------
        if (wid < 4) {
            int mta = wid >> 1, nta = wid & 1;
            f32x8 acc = zero8();
            for (int k0 = 0; k0 < 256; k0 += 32)
                acc = wmma_bf16(ld_a(qb + mta * 16 * 264, 264, k0),
                                ld_b(knb + nta * 16 * 264, 264, k0), acc);
            int g = lane >> 4, n = lane & 15;
#pragma unroll
            for (int r = 0; r < 8; ++r) {
                int ii = mta * 16 + r + 8 * g, jj = nta * 16 + n;
                attnb[ii * 36 + jj] = (__bf16)((jj <= ii) ? acc[r] : 0.f);
            }
        }
        __syncthreads();

        // --- out = q @ S + attn @ u_adj  (write to global) -------------------
        f32x8 oacc[2]; oacc[0] = zero8(); oacc[1] = zero8();
        for (int k0 = 0; k0 < 256; k0 += 32) {
            bf16x16 aq = ld_a(qb + mt * 16 * 264, 264, k0);
#pragma unroll
            for (int j = 0; j < 2; ++j)
                oacc[j] = wmma_bf16(aq, ld_b_kxn_f32(S, 132, (ntb + j) * 16, k0), oacc[j]);
        }
        {
            bf16x16 aat = ld_a(attnb + mt * 16 * 36, 36, 0);
#pragma unroll
            for (int j = 0; j < 2; ++j)
                oacc[j] = wmma_bf16(aat, ld_b_kxn(uab, 136, (ntb + j) * 16, 0), oacc[j]);
            int g = lane >> 4, n = lane & 15;
#pragma unroll
            for (int j = 0; j < 2; ++j)
#pragma unroll
                for (int r = 0; r < 8; ++r) {
                    int ii = mt * 16 + r + 8 * g;
                    int col = (ntb + j) * 16 + n;
                    dout[(size_t)(b * 4096 + l0 + ii) * 1024 + h * 256 + half * 128 + col] =
                        oacc[j][r];
                }
        }
        __syncthreads();

        // --- S += kn^T @ u_adj  (in-place on LDS S, C operand = S tile) ------
#pragma unroll
        for (int mi = 0; mi < 2; ++mi) {
            int mts = wid * 2 + mi;
            bf16x16 akt = ld_aT(knb, 264, mts * 16, 0);
#pragma unroll
            for (int nt = 0; nt < 8; ++nt) {
                float* st = S + (size_t)(mts * 16) * 132 + nt * 16;
                f32x8 c = ld_c_f32(st, 132);
                c = wmma_bf16(akt, ld_b_kxn(uab, 136, nt * 16, 0), c);
                st_c_f32(st, 132, c);
            }
        }
        __syncthreads();
    }
}

// ---------------------------------------------------------------------------
// gate_in row build: [hs (1024 bf16) | stats 16 | pad 16] per (b,l,h) row
// ---------------------------------------------------------------------------
__global__ __launch_bounds__(256) void gatein_kernel(const float* __restrict__ hs,
                                                     __bf16* __restrict__ g, int rows) {
    int row = blockIdx.x; int t = threadIdx.x;
    if (row >= rows) return;
    int bl = row >> 2;
    const float* xr = hs + (size_t)bl * 1024;
    __bf16* gr = g + (size_t)row * 1056;
#pragma unroll
    for (int e = 0; e < 4; ++e) { int c = t + e * 256; gr[c] = (__bf16)xr[c]; }
    if (t < 16) gr[1040 + t] = (__bf16)0.f;
}

// per-head stats: one wave per (row, tensor): mean, var(pop), abs-mean, l2
__global__ __launch_bounds__(256) void stats_kernel(
    const float* __restrict__ ls, const float* __restrict__ ll,
    const float* __restrict__ dd, const float* __restrict__ vv,
    __bf16* __restrict__ g, int jobs) {
    int j = (blockIdx.x * 256 + threadIdx.x) >> 5;
    int lane = threadIdx.x & 31;
    if (j >= jobs) return;
    int row = j >> 2, tix = j & 3;
    int bl = row >> 2, h = row & 3;
    const float* src = (tix == 0) ? ls : (tix == 1) ? ll : (tix == 2) ? dd : vv;
    const float* p = src + (size_t)bl * 1024 + h * 256;
    float s1 = 0.f, s2 = 0.f, sa = 0.f;
#pragma unroll
    for (int e = 0; e < 8; ++e) {
        float x = p[e * 32 + lane];
        s1 += x; s2 += x * x; sa += fabsf(x);
    }
#pragma unroll
    for (int m = 16; m >= 1; m >>= 1) {
        s1 += __shfl_xor(s1, m, 32);
        s2 += __shfl_xor(s2, m, 32);
        sa += __shfl_xor(sa, m, 32);
    }
    if (lane == 0) {
        float mean = s1 * (1.f / 256.f);
        float var  = s2 * (1.f / 256.f) - mean * mean;
        float am   = sa * (1.f / 256.f);
        float l2   = sqrtf(s2);
        __bf16* o = g + (size_t)row * 1056 + 1024 + tix * 4;
        o[0] = (__bf16)mean; o[1] = (__bf16)var; o[2] = (__bf16)am; o[3] = (__bf16)l2;
    }
}

// logits = hdn @ w2^T + b2 ; fw = softmax(logits / (softplus(lt)+1e-4))
__global__ __launch_bounds__(256) void gate_kernel(
    const __bf16* __restrict__ hdn, const float* __restrict__ w2,
    const float* __restrict__ b2, const float* __restrict__ lt,
    float* __restrict__ fw, int rows) {
    int j = (blockIdx.x * 256 + threadIdx.x) >> 5;
    int lane = threadIdx.x & 31;
    if (j >= rows) return;
    const __bf16* hr = hdn + (size_t)j * 1024;
    float lg[4];
#pragma unroll
    for (int o = 0; o < 4; ++o) {
        const float* wr = w2 + (size_t)o * 1024;
        float s = 0.f;
#pragma unroll
        for (int e = 0; e < 32; ++e) { int c = e * 32 + lane; s += (float)hr[c] * wr[c]; }
#pragma unroll
        for (int m = 16; m >= 1; m >>= 1) s += __shfl_xor(s, m, 32);
        lg[o] = s + b2[o];
    }
    if (lane == 0) {
        float temp = log1pf(expf(lt[0])) + 1e-4f;
        float mx = fmaxf(fmaxf(lg[0], lg[1]), fmaxf(lg[2], lg[3]));
        float es[4], se = 0.f;
#pragma unroll
        for (int o = 0; o < 4; ++o) { es[o] = expf((lg[o] - mx) / temp); se += es[o]; }
#pragma unroll
        for (int o = 0; o < 4; ++o) fw[(size_t)j * 4 + o] = es[o] / se;
    }
}

// combine + per-head RMS norm -> bf16 (input to final o_proj GEMM)
__global__ __launch_bounds__(256) void combine_kernel(
    const float* __restrict__ ls, const float* __restrict__ ll,
    const float* __restrict__ dd, const float* __restrict__ vv,
    const float* __restrict__ fw, const float* __restrict__ rss,
    const float* __restrict__ rsl, const float* __restrict__ onw,
    __bf16* __restrict__ omix, int jobs) {
    int j = (blockIdx.x * 256 + threadIdx.x) >> 5;
    int lane = threadIdx.x & 31;
    if (j >= jobs) return;
    int bl = j >> 2, h = j & 3;
    size_t base = (size_t)bl * 1024 + h * 256;
    const float* pf = fw + (size_t)j * 4;
    float f0 = pf[0], f1 = pf[1], f2 = pf[2], f3 = pf[3];
    float a = rss[0], bsc = rsl[0];
    float ov[8], s2 = 0.f;
#pragma unroll
    for (int e = 0; e < 8; ++e) {
        int d = e * 32 + lane;
        float o = (f0 + a) * ls[base + d] + (f1 + bsc) * ll[base + d] +
                  f2 * dd[base + d] + f3 * vv[base + d];
        ov[e] = o; s2 += o * o;
    }
#pragma unroll
    for (int m = 16; m >= 1; m >>= 1) s2 += __shfl_xor(s2, m, 32);
    float r = rsqrtf(s2 * (1.f / 256.f) + 1e-5f);
#pragma unroll
    for (int e = 0; e < 8; ++e) {
        int d = e * 32 + lane;
        omix[base + d] = (__bf16)(ov[e] * r * onw[d]);
    }
}

// ---------------------------------------------------------------------------
// Host driver
// ---------------------------------------------------------------------------
extern "C" void kernel_launch(void* const* d_in, const int* in_sizes, int n_in,
                              void* d_out, int out_size, void* d_ws, size_t ws_size,
                              hipStream_t stream) {
    (void)in_sizes; (void)n_in; (void)out_size; (void)ws_size;
    const int B = 2, L = 4096, D = 1024, H = 4;
    const int M = B * L;          // 8192
    const int MH = M * H;         // 32768
    const int KP = 1056;          // padded gate_in_dim

    const float* hs   = (const float*)d_in[0];
    const float* qw   = (const float*)d_in[1];
    const float* kw   = (const float*)d_in[2];
    const float* vw   = (const float*)d_in[3];
    const float* bw   = (const float*)d_in[4];
    const float* qcw  = (const float*)d_in[5];
    const float* kcw  = (const float*)d_in[6];
    const float* vcw  = (const float*)d_in[7];
    const float* firS = (const float*)d_in[8];
    const float* firL = (const float*)d_in[9];
    const float* w1   = (const float*)d_in[10];
    const float* b1   = (const float*)d_in[11];
    const float* w2   = (const float*)d_in[12];
    const float* b2   = (const float*)d_in[13];
    const float* lt   = (const float*)d_in[14];
    const float* rss  = (const float*)d_in[15];
    const float* rsl  = (const float*)d_in[16];
    const float* onw  = (const float*)d_in[17];
    const float* ow   = (const float*)d_in[18];

    char* ws = (char*)d_ws;
    size_t off = 0;
    auto take = [&](size_t bytes) -> char* {
        char* p = ws + off;
        off = (off + bytes + 255) & ~(size_t)255;
        return p;
    };

    __bf16* hsb  = (__bf16*)take((size_t)M * D * 2);
    __bf16* wqb  = (__bf16*)take((size_t)D * D * 2);
    __bf16* wkb  = (__bf16*)take((size_t)D * D * 2);
    __bf16* wvb  = (__bf16*)take((size_t)D * D * 2);
    __bf16* wob  = (__bf16*)take((size_t)D * D * 2);
    __bf16* w1b  = (__bf16*)take((size_t)D * KP * 2);
    char*   linR = take((size_t)3 * M * D * 4);  // q/k/v lin, later gate_in + omix
    float*  qlin = (float*)linR;
    float*  klin = qlin + (size_t)M * D;
    float*  vlin = klin + (size_t)M * D;
    float*  qc   = (float*)take((size_t)M * D * 4);
    float*  kc   = (float*)take((size_t)M * D * 4);
    float*  vc   = (float*)take((size_t)M * D * 4);
    float*  betab = (float*)take((size_t)MH * 4);
    float*  ddel = (float*)take((size_t)M * D * 4);
    float*  lsb  = (float*)take((size_t)M * D * 4);
    float*  llb  = (float*)take((size_t)M * D * 4);
    __bf16* hdnb = (__bf16*)take((size_t)MH * D * 2);
    float*  fwb  = (float*)take((size_t)MH * 4 * 4);
    // aliases into dead q/k/v-lin region (conv outputs consumed it already)
    __bf16* gin   = (__bf16*)linR;                                   // MH x 1056
    __bf16* omixb = (__bf16*)(linR + (((size_t)MH * KP * 2 + 255) & ~(size_t)255));

    const int n1 = M * D;  // 8388608

    // weight / activation converts
    cvt_bf16_kernel<<<n1 / 256, 256, 0, stream>>>(hs, hsb, n1);
    cvt_bf16_kernel<<<(D * D) / 256, 256, 0, stream>>>(qw, wqb, D * D);
    cvt_bf16_kernel<<<(D * D) / 256, 256, 0, stream>>>(kw, wkb, D * D);
    cvt_bf16_kernel<<<(D * D) / 256, 256, 0, stream>>>(vw, wvb, D * D);
    cvt_bf16_kernel<<<(D * D) / 256, 256, 0, stream>>>(ow, wob, D * D);
    cvt_w1_kernel<<<(D * KP) / 256, 256, 0, stream>>>(w1, w1b, D);

    // q/k/v projections (WMMA GEMM)
    dim3 g1(D / 128, M / 128);
    gemm_bf16_kernel<<<g1, 256, 0, stream>>>(hsb, wqb, qlin, nullptr, nullptr,
                                             M, D, D, D, D, D, 0);
    gemm_bf16_kernel<<<g1, 256, 0, stream>>>(hsb, wkb, klin, nullptr, nullptr,
                                             M, D, D, D, D, D, 0);
    gemm_bf16_kernel<<<g1, 256, 0, stream>>>(hsb, wvb, vlin, nullptr, nullptr,
                                             M, D, D, D, D, D, 0);

    // short conv + SiLU
    convsilu_kernel<<<n1 / 256, 256, 0, stream>>>(qlin, qcw, qc, n1);
    convsilu_kernel<<<n1 / 256, 256, 0, stream>>>(klin, kcw, kc, n1);
    convsilu_kernel<<<n1 / 256, 256, 0, stream>>>(vlin, vcw, vc, n1);

    // beta
    beta_kernel<<<(MH * 32) / 256, 256, 0, stream>>>(hs, bw, betab, MH);

    // delta rule chunkwise scan (LDS-resident state, WMMA throughout)
    delta_kernel<<<16, 256, 212352, stream>>>(qc, kc, vc, betab, ddel);

    // FIR branches on v_direct (L2-resident re-reads)
    fir_kernel<<<n1 / 256, 256, 0, stream>>>(vc, firS, lsb, 5, n1);
    fir_kernel<<<n1 / 256, 256, 0, stream>>>(vc, firL, llb, 64, n1);

    // gate input build + stats
    gatein_kernel<<<MH, 256, 0, stream>>>(hs, gin, MH);
    stats_kernel<<<(MH * 4 * 32) / 256, 256, 0, stream>>>(lsb, llb, ddel, vc, gin, MH * 4);

    // fuse MLP layer 1 (WMMA GEMM, fused bias + exact GeLU -> bf16)
    dim3 g2(D / 128, MH / 128);
    gemm_bf16_kernel<<<g2, 256, 0, stream>>>(gin, w1b, nullptr, hdnb, b1,
                                             MH, D, KP, KP, KP, D, 1);

    // logits + tempered softmax
    gate_kernel<<<(MH * 32) / 256, 256, 0, stream>>>(hdnb, w2, b2, lt, fwb, MH);

    // combine + per-head RMS norm
    combine_kernel<<<(MH * 32) / 256, 256, 0, stream>>>(lsb, llb, ddel, vc, fwb, rss, rsl,
                                                        onw, omixb, MH);

    // output projection (WMMA GEMM) -> d_out (f32)
    gemm_bf16_kernel<<<g1, 256, 0, stream>>>(omixb, wob, (float*)d_out, nullptr, nullptr,
                                             M, D, D, D, D, D, 0);
}
